// MuZeroNetFinal_39195871543606
// MI455X (gfx1250) — matrix operand order repaired and checked by reading, not compiled, not run on hardware
//
#include <hip/hip_runtime.h>
#include <math.h>

// ---------------------------------------------------------------------------
// Types for CDNA5 WMMA (wave32): bf16 A/B fragments, f32 accumulator.
// ---------------------------------------------------------------------------
typedef __attribute__((ext_vector_type(16))) __bf16 v16bf;
typedef __attribute__((ext_vector_type(8)))  float  v8f;

union V16U { v16bf v; __bf16 e[16]; };
union V8U  { v8f  v; float  e[8]; };

__device__ __forceinline__ float gelu_exact(float x) {
    // exact erf GELU (matches jax.nn.gelu(approximate=False))
    return 0.5f * x * (1.0f + erff(x * 0.7071067811865476f));
}

#define TM 128
#define TN 128
#define TK 32

// ---------------------------------------------------------------------------
// Generic GEMM: C[M,N] = act(A[M,K] @ W[K,N] + bias (+ resid))
// f32 in/out, bf16 WMMA compute (v_wmma_f32_16x16x32_bf16), f32 accumulate.
// Block: 256 threads = 8 waves; tile 128x128x32; each wave 32x64 (2x4 frags).
// Register-pipelined + LDS double-buffered: global loads of tile t+1 overlap
// the WMMAs of tile t; one s_barrier per K-tile.
// ---------------------------------------------------------------------------
__global__ __launch_bounds__(256)
void gemm_bf16_wmma(const float* __restrict__ A, const float* __restrict__ W,
                    const float* __restrict__ bias, const float* __restrict__ resid,
                    float* __restrict__ C, int M, int N, int K, int act)
{
    __shared__ __bf16 As[2][TM][TK + 8];   // [buf][m][k]
    __shared__ __bf16 Bs[2][TN][TK + 8];   // [buf][n][k] (transposed stage)

    const int tid  = threadIdx.x;
    const int lane = tid & 31;
    const int wid  = tid >> 5;
    const int wm   = wid >> 1;          // 0..3 : 32-row band
    const int wn   = wid & 1;           // 0..1 : 64-col band
    const int l15  = lane & 15;
    const int hh   = lane >> 4;         // lane half
    const int m0   = blockIdx.y * TM;
    const int n0   = blockIdx.x * TN;

    // fast path: fully interior tile, 16B-alignable rows
    const bool interior = (m0 + TM <= M) && (n0 + TN <= N) &&
                          ((K & 3) == 0) && ((N & 3) == 0);

    float4 ra[4], rbv[4];               // staged tile in registers

    auto load_tile = [&](int kt) {
        const bool fast = interior && (kt + TK <= K);
        if (fast) {
#pragma unroll
            for (int i = 0; i < 4; ++i) {           // A: 128x32, float4 along k
                int gi = tid + i * 256;
                int row = gi >> 3, c4 = (gi & 7) << 2;
                ra[i] = *reinterpret_cast<const float4*>(
                    A + (size_t)(m0 + row) * K + kt + c4);
            }
#pragma unroll
            for (int i = 0; i < 4; ++i) {           // B: 32x128, float4 along n
                int gi = tid + i * 256;
                int c = gi >> 5, n4 = (gi & 31) << 2;
                rbv[i] = *reinterpret_cast<const float4*>(
                    W + (size_t)(kt + c) * N + n0 + n4);
            }
        } else {
#pragma unroll
            for (int i = 0; i < 4; ++i) {
                int gi = tid + i * 256;
                int row = gi >> 3, c4 = (gi & 7) << 2;
                int gm = m0 + row;
                const float* ap = A + (size_t)gm * K + kt + c4;
                ra[i].x = (gm < M && kt + c4 + 0 < K) ? ap[0] : 0.0f;
                ra[i].y = (gm < M && kt + c4 + 1 < K) ? ap[1] : 0.0f;
                ra[i].z = (gm < M && kt + c4 + 2 < K) ? ap[2] : 0.0f;
                ra[i].w = (gm < M && kt + c4 + 3 < K) ? ap[3] : 0.0f;
            }
#pragma unroll
            for (int i = 0; i < 4; ++i) {
                int gi = tid + i * 256;
                int c = gi >> 5, n4 = (gi & 31) << 2;
                int gk = kt + c;
                const float* wp = W + (size_t)gk * N + n0 + n4;
                rbv[i].x = (gk < K && n0 + n4 + 0 < N) ? wp[0] : 0.0f;
                rbv[i].y = (gk < K && n0 + n4 + 1 < N) ? wp[1] : 0.0f;
                rbv[i].z = (gk < K && n0 + n4 + 2 < N) ? wp[2] : 0.0f;
                rbv[i].w = (gk < K && n0 + n4 + 3 < N) ? wp[3] : 0.0f;
            }
        }
    };

    union PK { __bf16 h[4]; unsigned long long u; };
    auto store_tile = [&](int buf) {
#pragma unroll
        for (int i = 0; i < 4; ++i) {               // A: packed ds_store_b64
            int gi = tid + i * 256;
            int row = gi >> 3, c4 = (gi & 7) << 2;
            PK p;
            p.h[0] = (__bf16)ra[i].x; p.h[1] = (__bf16)ra[i].y;
            p.h[2] = (__bf16)ra[i].z; p.h[3] = (__bf16)ra[i].w;
            *reinterpret_cast<unsigned long long*>(&As[buf][row][c4]) = p.u;
        }
#pragma unroll
        for (int i = 0; i < 4; ++i) {               // B: transpose into [n][k]
            int gi = tid + i * 256;
            int c = gi >> 5, n4 = (gi & 31) << 2;
            Bs[buf][n4 + 0][c] = (__bf16)rbv[i].x;
            Bs[buf][n4 + 1][c] = (__bf16)rbv[i].y;
            Bs[buf][n4 + 2][c] = (__bf16)rbv[i].z;
            Bs[buf][n4 + 3][c] = (__bf16)rbv[i].w;
        }
    };

    V8U acc[2][4];
#pragma unroll
    for (int f = 0; f < 2; ++f)
#pragma unroll
        for (int g = 0; g < 4; ++g)
#pragma unroll
            for (int r = 0; r < 8; ++r) acc[f][g].e[r] = 0.0f;

    auto compute = [&](int buf) {
        V16U fa[2], fb[4];
#pragma unroll
        for (int f = 0; f < 2; ++f) {
            int row = wm * 32 + f * 16 + l15;       // M index for this lane
#pragma unroll
            for (int p = 0; p < 8; ++p) {
                // ISA 16-bit A(16x32): VGPR p holds K = hh*8 + (p&3)*2 (+16 p>=4)
                int kb = hh * 8 + (p & 3) * 2 + ((p >> 2) << 4);
                fa[f].e[2 * p]     = As[buf][row][kb];
                fa[f].e[2 * p + 1] = As[buf][row][kb + 1];
            }
        }
#pragma unroll
        for (int g = 0; g < 4; ++g) {
            int col = wn * 64 + g * 16 + l15;       // N index for this lane
#pragma unroll
            for (int i2 = 0; i2 < 16; ++i2)
                fb[g].e[i2] = Bs[buf][col][hh * 8 + i2 + ((i2 >> 3) << 3)]; // K = hh*8 + i2 (+8 for i2>=8)
        }
#pragma unroll
        for (int f = 0; f < 2; ++f)
#pragma unroll
            for (int g = 0; g < 4; ++g)
                acc[f][g].v = __builtin_amdgcn_wmma_f32_16x16x32_bf16(
                    false, fa[f].v, false, fb[g].v,
                    (short)0, acc[f][g].v, false, false);
    };

    const int nt = (K + TK - 1) / TK;
    load_tile(0);
    for (int t = 0; t < nt; ++t) {
        store_tile(t & 1);
        __syncthreads();
        if (t + 1 < nt) load_tile((t + 1) * TK);   // overlap with WMMAs below
        compute(t & 1);
    }

    // ---- epilogue: C/D layout — lane(n = l15), VGPR r -> M = hh*8 + r ----
#pragma unroll
    for (int g = 0; g < 4; ++g) {
        int col = n0 + wn * 64 + g * 16 + l15;
        if (col >= N) continue;
        float bv = bias ? bias[col] : 0.0f;
#pragma unroll
        for (int f = 0; f < 2; ++f) {
#pragma unroll
            for (int r = 0; r < 8; ++r) {
                int row = m0 + wm * 32 + f * 16 + hh * 8 + r;
                if (row < M) {
                    float x = acc[f][g].e[r] + bv;
                    if (resid) x += resid[(size_t)row * N + col];
                    if (act == 1) x = gelu_exact(x);
                    C[(size_t)row * N + col] = x;
                }
            }
        }
    }
}

// ---------------------------------------------------------------------------
// LayerNorm: one block (256 threads) per row.
// ---------------------------------------------------------------------------
__global__ __launch_bounds__(256)
void layernorm_kernel(const float* __restrict__ X, const float* __restrict__ g,
                      const float* __restrict__ b, float* __restrict__ Y, int D)
{
    int row = blockIdx.x;
    const float* x = X + (size_t)row * D;
    float* y = Y + (size_t)row * D;
    __shared__ float s1[256], s2[256];
    float a = 0.0f, q = 0.0f;
    for (int i = threadIdx.x; i < D; i += 256) { float v = x[i]; a += v; q += v * v; }
    s1[threadIdx.x] = a; s2[threadIdx.x] = q;
    __syncthreads();
    for (int o = 128; o > 0; o >>= 1) {
        if (threadIdx.x < o) {
            s1[threadIdx.x] += s1[threadIdx.x + o];
            s2[threadIdx.x] += s2[threadIdx.x + o];
        }
        __syncthreads();
    }
    float mu  = s1[0] / (float)D;
    float var = s2[0] / (float)D - mu * mu;
    float inv = rsqrtf(var + 1e-5f);
    for (int i = threadIdx.x; i < D; i += 256)
        y[i] = (x[i] - mu) * inv * g[i] + b[i];
}

// ---------------------------------------------------------------------------
// MoE: softmax over 4 gate logits, top-2, renormalized weights (zeros elsewhere)
// ---------------------------------------------------------------------------
__global__ void moe_weights_kernel(const float* __restrict__ logits,
                                   float* __restrict__ Wout, int Brows)
{
    int b = blockIdx.x * blockDim.x + threadIdx.x;
    if (b >= Brows) return;
    float p[4];
    float mx = -1e30f;
    for (int e = 0; e < 4; ++e) mx = fmaxf(mx, logits[b * 4 + e]);
    float s = 0.0f;
    for (int e = 0; e < 4; ++e) { p[e] = expf(logits[b * 4 + e] - mx); s += p[e]; }
    for (int e = 0; e < 4; ++e) p[e] /= s;
    int i0 = 0;
    for (int e = 1; e < 4; ++e) if (p[e] > p[i0]) i0 = e;
    int i1 = -1;
    for (int e = 0; e < 4; ++e) { if (e == i0) continue; if (i1 < 0 || p[e] > p[i1]) i1 = e; }
    float ws2 = p[i0] + p[i1];
    for (int e = 0; e < 4; ++e)
        Wout[b * 4 + e] = (e == i0 || e == i1) ? p[e] / ws2 : 0.0f;
}

__global__ void scale_add_kernel(float* __restrict__ O, const float* __restrict__ T,
                                 const float* __restrict__ Wg, int e, int D, int n)
{
    int i = blockIdx.x * blockDim.x + threadIdx.x;
    if (i >= n) return;
    int b = i / D;
    O[i] += Wg[b * 4 + e] * T[i];
}

__global__ void tanh_kernel(float* X, int n)
{ int i = blockIdx.x * blockDim.x + threadIdx.x; if (i < n) X[i] = tanhf(X[i]); }

__global__ void sigmoid_kernel(float* X, int n)
{ int i = blockIdx.x * blockDim.x + threadIdx.x; if (i < n) X[i] = 1.0f / (1.0f + expf(-X[i])); }

// concat [shared(512) | phase(0.5,0.5,0.5)] -> [B,515]
__global__ void concat_phase_kernel(const float* __restrict__ S, float* __restrict__ O, int Brows)
{
    int i = blockIdx.x * blockDim.x + threadIdx.x;
    if (i >= Brows * 515) return;
    int b = i / 515, j = i % 515;
    O[i] = (j < 512) ? S[b * 512 + j] : 0.5f;
}

// value = softmax3(lg) . (ve, vm, vl)
__global__ void blend_value_kernel(const float* __restrict__ lg,
                                   const float* __restrict__ ve, const float* __restrict__ vm,
                                   const float* __restrict__ vl, float* __restrict__ outv, int Brows)
{
    int b = blockIdx.x * blockDim.x + threadIdx.x;
    if (b >= Brows) return;
    float a = lg[b * 3 + 0], c = lg[b * 3 + 1], d = lg[b * 3 + 2];
    float mx = fmaxf(a, fmaxf(c, d));
    float ea = expf(a - mx), eb = expf(c - mx), ec = expf(d - mx);
    float s = ea + eb + ec;
    outv[b] = (ea * ve[b] + eb * vm[b] + ec * vl[b]) / s;
}

// ---------------------------------------------------------------------------
// Host orchestration.
// Input leaf order: setup_inputs() dict insertion order, depth-first.
//   lin = {w,b}; lnp = {g,b}; rb = {ln1,fc1,ln2,fc2}; moe = {gate,w1,b1,w2,b2}
// ---------------------------------------------------------------------------
extern "C" void kernel_launch(void* const* d_in, const int* in_sizes, int n_in,
                              void* d_out, int out_size, void* d_ws, size_t ws_size,
                              hipStream_t stream)
{
    (void)in_sizes; (void)n_in; (void)out_size; (void)ws_size;
#define IN(i) ((const float*)d_in[(i)])
    const int B = 1024;
    float* out = (float*)d_out;
    float* ws  = (float*)d_ws;

    // workspace pools (~224 MB total)
    float* BIG0 = ws;                                 // B*16384
    float* BIG1 = BIG0 + (size_t)B * 16384;           // B*16384
    float* MED0 = BIG1 + (size_t)B * 16384;           // B*8192
    float* MED1 = MED0 + (size_t)B * 8192;            // B*8192
    float* SM   = MED1 + (size_t)B * 8192;            // B*8192 small pool
    float* SM0 = SM;                                  // B*2048
    float* SM1 = SM + (size_t)B * 2048;               // B*2048
    float* SM2 = SM + (size_t)B * 4096;               // B*2048
    float* SM3 = SM + (size_t)B * 6144;               // B*1024
    float* SM4 = SM + (size_t)B * 7168;               // B*1024
    float* GATE_L = SM3;
    float* GATE_W = SM3 + 8192;

    auto gemm = [&](const float* A, const float* W, const float* bias, const float* resid,
                    float* C, int M, int N, int K, int act) {
        dim3 g((N + TN - 1) / TN, (M + TM - 1) / TM);
        gemm_bf16_wmma<<<g, dim3(256), 0, stream>>>(A, W, bias, resid, C, M, N, K, act);
    };
    auto ln = [&](const float* X, const float* gam, const float* bet, float* Y, int D) {
        layernorm_kernel<<<dim3(B), dim3(256), 0, stream>>>(X, gam, bet, Y, D);
    };
    // pre-norm residual block: out = x + fc2(LN(gelu(fc1(LN(x)))))
    auto resblock = [&](const float* x, int d,
                        const float* l1g, const float* l1b,
                        const float* f1w, const float* f1b,
                        const float* l2g, const float* l2b,
                        const float* f2w, const float* f2b,
                        float* tmpA, float* tmpB, float* tmpC, float* outb) {
        ln(x, l1g, l1b, tmpA, d);
        gemm(tmpA, f1w, f1b, nullptr, tmpB, B, 2 * d, d, 1);
        ln(tmpB, l2g, l2b, tmpC, 2 * d);
        gemm(tmpC, f2w, f2b, x, outb, B, d, 2 * d, 0);
    };
    // dense top-2 MoE, accumulated expert-by-expert (saves a [B,E,D] buffer)
    auto moe = [&](const float* x, const float* gw, const float* gb,
                   const float* w1, const float* b1, const float* w2, const float* b2,
                   int D, int H, float* outb) {
        gemm(x, gw, gb, nullptr, GATE_L, B, 4, D, 0);
        moe_weights_kernel<<<dim3((B + 255) / 256), dim3(256), 0, stream>>>(GATE_L, GATE_W, B);
        hipMemcpyAsync(outb, x, (size_t)B * D * sizeof(float), hipMemcpyDeviceToDevice, stream);
        for (int e = 0; e < 4; ++e) {
            gemm(x, w1 + (size_t)e * D * H, b1 + (size_t)e * H, nullptr, BIG0, B, H, D, 1);
            gemm(BIG0, w2 + (size_t)e * H * D, b2 + (size_t)e * D, nullptr, BIG1, B, D, H, 0);
            scale_add_kernel<<<dim3((B * D + 255) / 256), dim3(256), 0, stream>>>(
                outb, BIG1, GATE_W, e, D, B * D);
        }
    };

    const float* obs = IN(0);

    // ---------------- representation trunk ----------------
    gemm(obs, IN(1), IN(2), nullptr, BIG0, B, 8192, 512, 1);        // lin0 + gelu
    ln(BIG0, IN(3), IN(4), MED1, 8192);                              // ln0 -> h
    resblock(MED1, 8192, IN(5), IN(6), IN(7), IN(8), IN(9), IN(10), IN(11), IN(12),
             MED0, BIG0, BIG1, MED0);                                // rb1 -> MED0
    gemm(MED0, IN(13), IN(14), nullptr, BIG0, B, 1536, 8192, 1);     // lin1 + gelu
    ln(BIG0, IN(15), IN(16), MED1, 1536);                            // ln1 -> h
    resblock(MED1, 1536, IN(17), IN(18), IN(19), IN(20), IN(21), IN(22), IN(23), IN(24),
             SM0, BIG0, BIG1, MED0);                                 // rb2 -> MED0
    gemm(MED0, IN(25), IN(26), nullptr, BIG0, B, 4096, 1536, 1);     // lin2 + gelu
    ln(BIG0, IN(27), IN(28), MED1, 4096);                            // ln2 -> h
    resblock(MED1, 4096, IN(29), IN(30), IN(31), IN(32), IN(33), IN(34), IN(35), IN(36),
             MED0, BIG0, BIG1, MED0);                                // rb3 -> MED0
    moe(MED0, IN(37), IN(38), IN(39), IN(40), IN(41), IN(42), 4096, 1536, MED1); // rep moe
    resblock(MED1, 4096, IN(43), IN(44), IN(45), IN(46), IN(47), IN(48), IN(49), IN(50),
             MED0, BIG0, BIG1, MED0);                                // rb4 -> latent h = MED0

    // latent h -> output slot 10
    hipMemcpyAsync(out + 134144, MED0, (size_t)B * 4096 * sizeof(float),
                   hipMemcpyDeviceToDevice, stream);

    // ---------------- policy ----------------
    moe(MED0, IN(51), IN(52), IN(53), IN(54), IN(55), IN(56), 4096, 1024, MED1); // pl = MED1
    {
        const int hd[4] = {64, 32, 16, 8};
        const size_t pofs[4] = {0, 65536, 98304, 114688};
        for (int i = 0; i < 4; ++i) {
            int bi = 57 + 10 * i;
            gemm(MED1, IN(bi + 0), IN(bi + 1), nullptr, SM0, B, 512, 4096, 1); // l1 + gelu
            ln(SM0, IN(bi + 2), IN(bi + 3), SM1, 512);                          // n1
            gemm(SM1, IN(bi + 4), IN(bi + 5), nullptr, SM2, B, 256, 512, 1);    // l2 + gelu
            ln(SM2, IN(bi + 6), IN(bi + 7), SM3, 256);                          // n2
            gemm(SM3, IN(bi + 8), IN(bi + 9), nullptr, out + pofs[i], B, hd[i], 256, 0); // l3
        }
    }

    // ---------------- value ----------------
    gemm(MED0, IN(97), IN(98), nullptr, SM0, B, 512, 4096, 1);       // trunk + gelu
    ln(SM0, IN(99), IN(100), SM1, 512);                              // shared
    gemm(SM1, IN(101), IN(102), nullptr, SM2, B, 256, 512, 1);       // early
    gemm(SM2, IN(103), IN(104), nullptr, SM4 + 0, B, 1, 256, 0);
    gemm(SM1, IN(105), IN(106), nullptr, SM2, B, 256, 512, 1);       // mid
    gemm(SM2, IN(107), IN(108), nullptr, SM4 + B, B, 1, 256, 0);
    gemm(SM1, IN(109), IN(110), nullptr, SM2, B, 256, 512, 1);       // late
    gemm(SM2, IN(111), IN(112), nullptr, SM4 + 2 * B, B, 1, 256, 0);
    tanh_kernel<<<dim3((3 * B + 255) / 256), dim3(256), 0, stream>>>(SM4, 3 * B);
    concat_phase_kernel<<<dim3((B * 515 + 255) / 256), dim3(256), 0, stream>>>(SM1, BIG0, B);
    gemm(BIG0, IN(113), IN(114), nullptr, SM2, B, 64, 515, 1);       // blend l1 + gelu
    gemm(SM2, IN(115), IN(116), nullptr, SM4 + 3 * B, B, 3, 64, 0);  // blend logits
    blend_value_kernel<<<dim3((B + 255) / 256), dim3(256), 0, stream>>>(
        SM4 + 3 * B, SM4, SM4 + B, SM4 + 2 * B, out + 122880, B);

    // ---------------- aux heads ----------------
    gemm(MED0, IN(117), IN(118), nullptr, SM0, B, 512, 4096, 1);     // feat = gelu(trunk)
    gemm(SM0, IN(119), IN(120), nullptr, SM2, B, 128, 512, 1);       // tr
    gemm(SM2, IN(121), IN(122), nullptr, out + 123904, B, 1, 128, 0);
    gemm(SM0, IN(123), IN(124), nullptr, SM2, B, 128, 512, 1);       // prod
    gemm(SM2, IN(125), IN(126), nullptr, out + 124928, B, 6, 128, 0);
    gemm(SM0, IN(127), IN(128), nullptr, SM2, B, 128, 512, 1);       // vp
    gemm(SM2, IN(129), IN(130), nullptr, out + 131072, B, 1, 128, 0);
    gemm(SM0, IN(131), IN(132), nullptr, SM2, B, 64, 512, 1);        // cp
    gemm(SM2, IN(133), IN(134), nullptr, out + 132096, B, 1, 64, 0);
    sigmoid_kernel<<<dim3((B + 255) / 256), dim3(256), 0, stream>>>(out + 132096, B);
    gemm(SM0, IN(135), IN(136), nullptr, SM2, B, 64, 512, 1);        // rs
    gemm(SM2, IN(137), IN(138), nullptr, out + 133120, B, 1, 64, 0);
    sigmoid_kernel<<<dim3((B + 255) / 256), dim3(256), 0, stream>>>(out + 133120, B);
#undef IN
}
